// RNN_MultiRegional_D1D2_30356828848602
// MI455X (gfx1250) — compile-verified, matrix operand order checked by reading, not run on hardware
//
#include <hip/hip_runtime.h>
#include <hip/hip_bf16.h>

typedef __attribute__((ext_vector_type(16))) __bf16 v16bf;
typedef __attribute__((ext_vector_type(8)))  __bf16 v8bf;
typedef __attribute__((ext_vector_type(8)))  float  v8f;
typedef __attribute__((ext_vector_type(4)))  unsigned int v4u;
typedef __attribute__((ext_vector_type(8)))  int v8i;
typedef __attribute__((ext_vector_type(4)))  int v4i;

#define HDIM   512
#define SIX    3072
#define BATCH  32
#define TSTEPS 1000
#define KT     (SIX / 32)    // 96 k-tiles
#define NTILES (SIX / 16)    // 192 n-tiles
#define DTC    0.01f
#define NIH    359           // H - int(0.3*H) = 512 - 153
#define SLABW  256           // K-columns staged per TDM slab
#define NSLAB  (SIX / SLABW) // 12
#define KT_PER_SLAB (SLABW / 32) // 8

__device__ __forceinline__ float ht(float w) {
    return fminf(fmaxf(w, 1e-15f), 1.0f);
}

// ---------------------------------------------------------------------------
// Kernel 1: build effective W_rec into a WMMA B-fragment-packed bf16 layout.
// Tile (nt, kt) at element offset (nt*KT + kt)*512; lane l holds 16 contiguous
// bf16 at l*16: n = l&15, kh = l>>4, elem e -> k = (e&7) + kh*8 + (e>=8)*16.
// ---------------------------------------------------------------------------
__global__ void build_weights(
    const float* __restrict__ str2str_fixed,
    const float* __restrict__ str2gpe_w,  const float* __restrict__ gpe2stn_w,
    const float* __restrict__ stn2snr_w,  const float* __restrict__ str2snr_w,
    const float* __restrict__ snr2thal_w, const float* __restrict__ thal2alm_w,
    const float* __restrict__ thal2str_w, const float* __restrict__ alm2alm_w,
    const float* __restrict__ alm2str_w,
    __bf16* __restrict__ Wp)
{
    int idx = blockIdx.x * blockDim.x + threadIdx.x;
    if (idx >= SIX * SIX) return;
    int k = idx % SIX;          // K index = col of W_rec
    int j = idx / SIX;          // N index = row of W_rec
    int r = j >> 9, c = k >> 9;
    int jr = j & 511, kc = k & 511;
    int wi = jr * HDIM + kc;

    float w = 0.0f;
    if (r == 0) {                                     // -> STR
        if (c == 0)      w = -str2str_fixed[wi];
        else if (c == 4) w = (jr >= 128 && jr < 384) ? ht(thal2str_w[wi]) : 0.0f;
        else if (c == 5) w = (kc < NIH) ? ht(alm2str_w[wi]) : 0.0f;
    } else if (r == 1) {                              // -> GPe
        if (c == 0)      w = (kc >= 256) ? -ht(str2gpe_w[wi]) : 0.0f;
    } else if (r == 2) {                              // -> STN
        if (c == 1)      w = -ht(gpe2stn_w[wi]);
    } else if (r == 3) {                              // -> SNr
        if (c == 0)      w = (kc < 256) ? -ht(str2snr_w[wi]) : 0.0f;
        else if (c == 2) w = ht(stn2snr_w[wi]);
    } else if (r == 4) {                              // -> THAL
        if (c == 3)      w = -ht(snr2thal_w[wi]);
    } else {                                          // -> ALM
        if (c == 4)      w = ht(thal2alm_w[wi]);
        else if (c == 5) w = ht(alm2alm_w[wi]) * ((kc >= NIH) ? -1.0f : 1.0f);
    }

    int kt  = k >> 5, kin = k & 31;
    int nt  = j >> 4, nin = j & 15;
    int kh  = (kin >> 3) & 1;
    int e   = (kin & 7) + ((kin >> 4) << 3);
    int lane = kh * 16 + nin;
    size_t off = ((size_t)(nt * KT + kt)) * 512 + (size_t)lane * 16 + e;
    Wp[off] = (__bf16)w;
}

// ---------------------------------------------------------------------------
// Kernel 2: masked input weights, tonic vector, h0 in fp32 + bf16.
// ---------------------------------------------------------------------------
__global__ void prep_state(
    const float* __restrict__ hn, const float* __restrict__ inp_weight,
    float* __restrict__ h0f, __bf16* __restrict__ h0b,
    float* __restrict__ iw_eff, float* __restrict__ tonic)
{
    int idx = blockIdx.x * blockDim.x + threadIdx.x;
    if (idx < SIX) {
        int n = idx;
        float sm = (n >= 128 && n < 384) ? 1.0f : 0.0f;
        iw_eff[n] = ht(inp_weight[n]) * sm;
        tonic[n]  = ((n >= 512 && n < 1536) || (n >= 2048 && n < 2560)) ? 1.0f : 0.0f;
    }
    if (idx < BATCH * SIX) {
        float v = hn[idx];
        h0f[idx] = v;
        h0b[idx] = (__bf16)v;
    }
}

// ---------------------------------------------------------------------------
// TDM: issue one tensor_load_to_lds for a [32 rows x SLABW cols] bf16 slab of
// the row-major h matrix (row stride SIX) into LDS at lds_off.
// D# per CDNA5 ISA: group0 = {count=1 | lds_addr |
// global_addr | type=2}; group1 = {data_size=2B, tensor_dim0=SLABW,
// tensor_dim1=32, tile_dim0=SLABW, tile_dim1=32, dim0_stride=SIX}.
// ---------------------------------------------------------------------------
__device__ __forceinline__ void tdm_load_slab(const __bf16* gsrc, unsigned lds_off) {
    unsigned long long ga = (unsigned long long)(size_t)gsrc;
    v4u g0 = { 1u,                                        // count=1, user desc
               lds_off,                                    // lds_addr
               (unsigned)(ga & 0xFFFFFFFFu),               // global_addr[31:0]
               (unsigned)((ga >> 32) & 0x1FFFFFFu) | (2u << 30) }; // [56:32] | type=2
    v8i g1 = { (int)(1u << 16),                 // data_size = 1 -> 2 bytes
               (int)((unsigned)SLABW << 16),    // tensor_dim0[15:0] @ [63:48]
               (int)(32u << 16),                // tensor_dim0 hi=0 | tensor_dim1=32
               (int)((unsigned)SLABW << 16),    // tensor_dim1 hi=0 | tile_dim0
               32,                              // tile_dim1=32, tile_dim2=0
               (int)SIX,                        // tensor_dim0_stride low32
               0, 0 };
    v4i g2 = { 0, 0, 0, 0 };
    v4i g3 = { 0, 0, 0, 0 };
#if __clang_major__ >= 23
    v8i g4 = { 0, 0, 0, 0, 0, 0, 0, 0 };
    __builtin_amdgcn_tensor_load_to_lds(g0, g1, g2, g3, g4, 0);
#else
    __builtin_amdgcn_tensor_load_to_lds(g0, g1, g2, g3, 0);
#endif
}

// ---------------------------------------------------------------------------
// Kernel 3: one scan step. 48 blocks x 4 waves; wave nt owns output columns
// [nt*16, nt*16+16) for all 32 batch rows. The A panel (h) is staged in LDS
// by the Tensor Data Mover in double-buffered 16KB slabs (s_wait_tensorcnt +
// barrier), A-fragments via ds_load_b128, B streamed from L2-resident packed
// weights, v_wmma_f32_16x16x32_bf16 with fp32 accumulation, fused
// leaky-integrate + relu + drive epilogue.
// ---------------------------------------------------------------------------
__global__ __launch_bounds__(128, 1)
void rnn_step(const __bf16* __restrict__ Wp,
              const __bf16* __restrict__ h_bf,    // prev h, bf16 row-major [32][3072]
              const float*  __restrict__ h_f,     // prev h, fp32 row-major
              float* __restrict__ hn_f, __bf16* __restrict__ hn_bf,
              const float* __restrict__ inp,      // [32][1000]
              const float* __restrict__ inhib,    // [32][1000][3072]
              const float* __restrict__ iw_eff,
              const float* __restrict__ tonic,
              float* __restrict__ out, int t)
{
    __shared__ __align__(128) __bf16 smem[2][BATCH * SLABW];   // 2 x 16KB... (32*256*2B = 16KB) x2

    const int lane = threadIdx.x & 31;
    const int wave = threadIdx.x >> 5;
    const int nt   = blockIdx.x * 4 + wave;       // 0..191
    const int n15  = lane & 15;
    const int kh   = lane >> 4;

    // scalar (SGPR) wave-0 predicate: TDM ignores EXEC, so guard with a
    // uniform branch, not a lane mask.
    const bool wave0 = (__builtin_amdgcn_readfirstlane((int)threadIdx.x) < 32);

    const unsigned lds0 = (unsigned)(size_t)&smem[0][0];
    const unsigned lds1 = (unsigned)(size_t)&smem[1][0];

    if (wave0) tdm_load_slab(h_bf, lds0);

    v8f c0 = {};
    v8f c1 = {};
    const __bf16* wt = Wp + (size_t)nt * KT * 512 + (size_t)lane * 16;

    for (int s = 0; s < NSLAB; ++s) {
        if (wave0) __builtin_amdgcn_s_wait_tensorcnt(0);
        __syncthreads();                       // slab s visible to all waves
        if (wave0 && (s + 1) < NSLAB)          // overlap: DMA slab s+1 into other buffer
            tdm_load_slab(h_bf + (size_t)(s + 1) * SLABW,
                          ((s + 1) & 1) ? lds1 : lds0);

        const __bf16* As  = (s & 1) ? &smem[1][0] : &smem[0][0];
        const __bf16* wts = wt + (size_t)s * KT_PER_SLAB * 512;

        #pragma unroll
        for (int kt = 0; kt < KT_PER_SLAB; ++kt) {
            v16bf b = *reinterpret_cast<const v16bf*>(wts + (size_t)kt * 512);
            if (kt + 2 < KT_PER_SLAB)
                __builtin_prefetch(wts + (size_t)(kt + 2) * 512, 0, 1);
            const __bf16* pa = As + n15 * SLABW + kt * 32 + kh * 8;   // rows 0..15
            const __bf16* pb = pa + 16 * SLABW;                       // rows 16..31
            v8bf a0lo = *reinterpret_cast<const v8bf*>(pa);
            v8bf a0hi = *reinterpret_cast<const v8bf*>(pa + 16);
            v8bf a1lo = *reinterpret_cast<const v8bf*>(pb);
            v8bf a1hi = *reinterpret_cast<const v8bf*>(pb + 16);
            v16bf a0 = __builtin_shufflevector(a0lo, a0hi,
                         0,1,2,3,4,5,6,7,8,9,10,11,12,13,14,15);
            v16bf a1 = __builtin_shufflevector(a1lo, a1hi,
                         0,1,2,3,4,5,6,7,8,9,10,11,12,13,14,15);
            c0 = __builtin_amdgcn_wmma_f32_16x16x32_bf16(
                     false, a0, false, b, (short)0, c0, false, false);
            c1 = __builtin_amdgcn_wmma_f32_16x16x32_bf16(
                     false, a1, false, b, (short)0, c1, false, false);
        }
    }

    const int   n   = nt * 16 + n15;
    const float iwn = iw_eff[n];
    const float ton = tonic[n];

    #pragma unroll
    for (int v = 0; v < 8; ++v) {
        // C/D layout: VGPR v, lanes 0-15 -> M=v, lanes 16-31 -> M=v+8; N=lane&15
        {
            int m = v + (kh << 3);
            size_t hidx = (size_t)m * SIX + n;
            size_t bt   = (size_t)m * TSTEPS + t;
            float hp  = h_f[hidx];
            float d   = inp[bt] * iwn + inhib[bt * SIX + n] + ton;
            float val = fmaxf(hp + DTC * (-hp + c0[v] + d), 0.0f);
            hn_f[hidx]  = val;
            hn_bf[hidx] = (__bf16)val;
            out[bt * SIX + n] = val;
        }
        {
            int m = v + (kh << 3) + 16;
            size_t hidx = (size_t)m * SIX + n;
            size_t bt   = (size_t)m * TSTEPS + t;
            float hp  = h_f[hidx];
            float d   = inp[bt] * iwn + inhib[bt * SIX + n] + ton;
            float val = fmaxf(hp + DTC * (-hp + c1[v] + d), 0.0f);
            hn_f[hidx]  = val;
            hn_bf[hidx] = (__bf16)val;
            out[bt * SIX + n] = val;
        }
    }
}

// ---------------------------------------------------------------------------
extern "C" void kernel_launch(void* const* d_in, const int* in_sizes, int n_in,
                              void* d_out, int out_size, void* d_ws, size_t ws_size,
                              hipStream_t stream) {
    (void)in_sizes; (void)n_in; (void)out_size; (void)ws_size;

    const float* inp        = (const float*)d_in[0];   // [32,1000,1]
    const float* hn         = (const float*)d_in[1];   // [1,32,3072]
    const float* inhib      = (const float*)d_in[3];   // [32,1000,3072]
    const float* str2gpe_w  = (const float*)d_in[5];
    const float* gpe2stn_w  = (const float*)d_in[6];
    const float* stn2snr_w  = (const float*)d_in[7];
    const float* str2snr_w  = (const float*)d_in[8];
    const float* snr2thal_w = (const float*)d_in[9];
    const float* thal2alm_w = (const float*)d_in[10];
    const float* thal2str_w = (const float*)d_in[11];
    const float* alm2alm_w  = (const float*)d_in[12];
    const float* alm2str_w  = (const float*)d_in[13];
    const float* inp_weight = (const float*)d_in[14];  // [1,3072]
    const float* str2str_fx = (const float*)d_in[15];

    char* ws = (char*)d_ws;
    size_t o = 0;
    __bf16* Wp = (__bf16*)(ws + o); o += (size_t)SIX * SIX * sizeof(__bf16);  // 18.9 MB
    float*  hf[2];
    hf[0] = (float*)(ws + o);  o += (size_t)BATCH * SIX * sizeof(float);
    hf[1] = (float*)(ws + o);  o += (size_t)BATCH * SIX * sizeof(float);
    __bf16* hb[2];
    hb[0] = (__bf16*)(ws + o); o += (size_t)BATCH * SIX * sizeof(__bf16);
    hb[1] = (__bf16*)(ws + o); o += (size_t)BATCH * SIX * sizeof(__bf16);
    float* iw_eff = (float*)(ws + o); o += SIX * sizeof(float);
    float* tonic  = (float*)(ws + o); o += SIX * sizeof(float);

    build_weights<<<(SIX * SIX + 255) / 256, 256, 0, stream>>>(
        str2str_fx, str2gpe_w, gpe2stn_w, stn2snr_w, str2snr_w,
        snr2thal_w, thal2alm_w, thal2str_w, alm2alm_w, alm2str_w, Wp);

    prep_state<<<(BATCH * SIX + 255) / 256, 256, 0, stream>>>(
        hn, inp_weight, hf[0], hb[0], iw_eff, tonic);

    float* out = (float*)d_out;
    for (int t = 0; t < TSTEPS; ++t) {
        int p = t & 1;
        rnn_step<<<NTILES / 4, 128, 0, stream>>>(
            Wp, hb[p], hf[p], hf[p ^ 1], hb[p ^ 1],
            inp, inhib, iw_eff, tonic, out, t);
    }
}
